// S4ConditionalSideChainModel_11038065951391
// MI455X (gfx1250) — matrix operand order repaired and checked by reading, not compiled, not run on hardware
//
#include <hip/hip_runtime.h>
#include <math.h>

// Problem constants (match reference)
#define Bn 16
#define Ln 16384
#define Hn 64
#define Nn 16
#define DEPTHn 4
#define MLP_HIDn 32
#define Tn 64            // chunk length
#define Jn (Ln / Tn)     // 256 chunks
#define LDSP 68          // padded LDS row stride (floats), conflict-free

typedef __attribute__((ext_vector_type(16))) _Float16 v16h;
typedef __attribute__((ext_vector_type(8)))  float    v8f;

// Branch-free tanh-approx gelu: 0.5*x*(1+tanh(y)) == x*sigmoid(2y)
//   sigmoid(2y) = rcp(1 + exp2(-2y*log2e)) : one v_exp_f32 + one v_rcp_f32.
__device__ __forceinline__ float gelu_f(float x) {
    const float c  = 1.5957691216057308f;   // 2*sqrt(2/pi)
    const float k1 = 0.044715f;
    const float l2e = 1.4426950408889634f;
    float y2 = c * (x + k1 * x * x * x);    // 2*y
    float e  = __builtin_amdgcn_exp2f(-l2e * y2);
    return x * __builtin_amdgcn_rcpf(1.0f + e);
}

// ---------------------------------------------------------------------------
// cond = 4-layer gelu MLP on control params: (16,2)->(16,32)->...  one block.
// ---------------------------------------------------------------------------
__global__ void cond_mlp_kernel(const float* __restrict__ cp,
                                const float* __restrict__ W0,
                                const float* __restrict__ b0,
                                const float* __restrict__ W,
                                const float* __restrict__ bb,
                                float* __restrict__ cond) {
    __shared__ float buf[2][Bn * MLP_HIDn];
    int tid = threadIdx.x;                 // 0..511
    int b = tid >> 5, j = tid & 31;
    float a = b0[j] + cp[b * 2 + 0] * W0[0 * 32 + j] + cp[b * 2 + 1] * W0[1 * 32 + j];
    buf[0][b * 32 + j] = gelu_f(a);
    __syncthreads();
    int cur = 0;
    for (int layer = 0; layer < 3; ++layer) {
        float s = bb[layer * 32 + j];
        for (int k = 0; k < 32; ++k)
            s += buf[cur][b * 32 + k] * W[(layer * 32 + k) * 32 + j];
        buf[1 - cur][b * 32 + j] = gelu_f(s);
        cur = 1 - cur;
        __syncthreads();
    }
    cond[b * 32 + j] = buf[cur][b * 32 + j];
}

// ---------------------------------------------------------------------------
// h[b,c,l] = x[b,l] * exp_W[c] + exp_b[c]     (B,H,L) f32 layout, 64 MiB
// ---------------------------------------------------------------------------
__global__ void expand_kernel(const float* __restrict__ x,
                              const float* __restrict__ exp_W,
                              const float* __restrict__ exp_b,
                              float* __restrict__ hbuf) {
    size_t idx = (size_t)blockIdx.x * blockDim.x + threadIdx.x; // B*H*L
    int l  = (int)(idx & (Ln - 1));
    int bh = (int)(idx >> 14);
    int hc = bh & 63;
    int b  = bh >> 6;
    hbuf[idx] = x[(size_t)b * Ln + l] * exp_W[hc] + exp_b[hc];
}

// ---------------------------------------------------------------------------
// Per-depth S4D coefficient prep.  Block = one channel h, 64 threads = t rows.
//   Kf[h][t]        f16 : K[h,t] = 2 Re sum_n c_n w_n^t            (Toeplitz)
//   Across[h][t][k] f16 : k<16 -> 2Re(c_n w_n^{t+1}), k>=16 -> -2Im(...)
//   wbuf[h][n], w64buf[h][n] : w = exp(dt*A), w^64  (for summary / scan)
// ---------------------------------------------------------------------------
__global__ void s4_prep_kernel(const float* __restrict__ log_dt,
                               const float* __restrict__ A_real,
                               const float* __restrict__ A_imag,
                               const float* __restrict__ C_re,
                               const float* __restrict__ C_im,
                               int d,
                               _Float16* __restrict__ Kf,
                               _Float16* __restrict__ Across,
                               float2* __restrict__ wbuf,
                               float2* __restrict__ w64buf) {
    int hc = blockIdx.x;
    int t  = threadIdx.x;                  // 0..63
    float dt = expf(log_dt[d * Hn + hc]);
    float ksum = 0.0f;
    for (int n = 0; n < Nn; ++n) {
        int base = (d * Hn + hc) * Nn + n;
        float Are = -expf(A_real[base]);
        float Aim = A_imag[base];
        float dre = dt * Are, dim = dt * Aim;
        float em  = expf(dre);
        float ere = em * cosf(dim), eim = em * sinf(dim);   // w = exp(dtA)
        float nre = ere - 1.0f, nim = eim;                   // exp(dtA)-1
        float cre0 = C_re[base], cim0 = C_im[base];
        float tre = cre0 * nre - cim0 * nim;                 // C0*(exp(dtA)-1)
        float tim = cre0 * nim + cim0 * nre;
        float den = Are * Are + Aim * Aim;
        float cre = (tre * Are + tim * Aim) / den;           // / A
        float cim = (tim * Are - tre * Aim) / den;
        float emt = expf(dre * (float)t);
        float ang = dim * (float)t;
        float etre = emt * cosf(ang), etim = emt * sinf(ang);
        ksum += 2.0f * (cre * etre - cim * etim);
        float emt1 = expf(dre * (float)(t + 1));
        float ang1 = dim * (float)(t + 1);
        float e1re = emt1 * cosf(ang1), e1im = emt1 * sinf(ang1);
        float pre = cre * e1re - cim * e1im;
        float pim = cre * e1im + cim * e1re;
        Across[((size_t)hc * Tn + t) * 32 + n]      = (_Float16)( 2.0f * pre);
        Across[((size_t)hc * Tn + t) * 32 + 16 + n] = (_Float16)(-2.0f * pim);
        if (t == n) {
            wbuf[hc * Nn + n] = make_float2(ere, eim);
            float em64 = expf(64.0f * dre);
            float a64  = 64.0f * dim;
            w64buf[hc * Nn + n] = make_float2(em64 * cosf(a64), em64 * sinf(a64));
        }
    }
    Kf[hc * Tn + t] = (_Float16)ksum;
}

// ---------------------------------------------------------------------------
// Pack the three A-matrices into WMMA A fragments (16-bit A 16x32, contiguous k order, same as B):
//   lane<16: K in {0..15}; lane>=16: K in {16..31}; m = lane&15.
// fragC/T0/T1 [h][tile mi][lane] -> v16h, so s4_apply loads one v16h per A.
// ---------------------------------------------------------------------------
__global__ void frag_build_kernel(const _Float16* __restrict__ Kf,
                                  const _Float16* __restrict__ Across,
                                  v16h* __restrict__ fragC,
                                  v16h* __restrict__ fragT0,
                                  v16h* __restrict__ fragT1) {
    int hc  = blockIdx.x;                  // 0..63
    int tid = threadIdx.x;                 // 0..127
    int lane = tid & 31, mi = tid >> 5;
    int t0 = mi * 16 + (lane & 15);
    int kbase = (lane < 16) ? 0 : 16;
    const _Float16* Ap = Across + ((size_t)hc * Tn + t0) * 32;
    const _Float16* Kp = Kf + hc * Tn;
    v16h c, a0, a1;
    #pragma unroll
    for (int j2 = 0; j2 < 16; ++j2) {
        int k = kbase + j2;
        c[j2]  = Ap[k];
        a0[j2] = (t0 >= k)      ? Kp[t0 - k]      : (_Float16)0.0f;
        a1[j2] = (t0 >= k + 32) ? Kp[t0 - k - 32] : (_Float16)0.0f;
    }
    int o = (hc * 4 + mi) * 32 + lane;
    fragC[o] = c; fragT0[o] = a0; fragT1[o] = a1;
}

// ---------------------------------------------------------------------------
// FiLM: gb = cond @ film_W[d] + film_b[d];  gamma=gb[:, :H], beta=gb[:, H:]
// ---------------------------------------------------------------------------
__global__ void film_kernel(const float* __restrict__ cond,
                            const float* __restrict__ film_W,
                            const float* __restrict__ film_b,
                            int d,
                            float* __restrict__ gamma,
                            float* __restrict__ beta) {
    int idx = blockIdx.x * blockDim.x + threadIdx.x; // B*2H = 2048
    if (idx >= Bn * 2 * Hn) return;
    int b = idx >> 7, col = idx & 127;
    float a = film_b[d * 2 * Hn + col];
    for (int k = 0; k < 32; ++k)
        a += cond[b * 32 + k] * film_W[((size_t)d * 32 + k) * (2 * Hn) + col];
    if (col < Hn) gamma[b * Hn + col] = a;
    else          beta[b * Hn + col - Hn] = a;
}

// ---------------------------------------------------------------------------
// Per-chunk input summary: one thread per (j,b,h) carries all N=16 complex
// states in registers: Sin[j,b,h,n] = sum_t w_n^{63-t} u[b,h,jT+t].
// ---------------------------------------------------------------------------
__global__ void chunk_summary_kernel(const float* __restrict__ hbuf,
                                     const float2* __restrict__ wbuf,
                                     float2* __restrict__ Sin) {
    int idx = blockIdx.x * blockDim.x + threadIdx.x; // J*B*H = 262144
    int hc = idx & 63;
    int b  = (idx >> 6) & 15;
    int j  = idx >> 10;
    float wr[Nn], wi[Nn], sr[Nn], si[Nn];
    #pragma unroll
    for (int n = 0; n < Nn; ++n) {
        float2 w = wbuf[hc * Nn + n];
        wr[n] = w.x; wi[n] = w.y; sr[n] = 0.0f; si[n] = 0.0f;
    }
    const float4* up = (const float4*)(hbuf + ((size_t)b * Hn + hc) * Ln + (size_t)j * Tn);
    for (int t4 = 0; t4 < Tn / 4; ++t4) {
        float4 u4 = up[t4];
        float uu[4] = {u4.x, u4.y, u4.z, u4.w};
        #pragma unroll
        for (int q = 0; q < 4; ++q) {
            float u = uu[q];
            #pragma unroll
            for (int n = 0; n < Nn; ++n) {
                float r = wr[n] * sr[n] - wi[n] * si[n] + u;
                si[n]   = wr[n] * si[n] + wi[n] * sr[n];
                sr[n]   = r;
            }
        }
    }
    float2* sp = Sin + (size_t)idx * Nn;   // idx == (j*B + b)*H + hc
    #pragma unroll
    for (int n = 0; n < Nn; ++n) sp[n] = make_float2(sr[n], si[n]);
}

// ---------------------------------------------------------------------------
// Sequential scan over the 256 chunks (in place: Sin -> boundary state Sb).
// ---------------------------------------------------------------------------
__global__ void chunk_scan_kernel(const float2* __restrict__ w64buf,
                                  float2* __restrict__ S) {
    int idx = blockIdx.x * blockDim.x + threadIdx.x; // B*H*N = 16384
    int n  = idx & 15;
    int hc = (idx >> 4) & 63;
    float2 w = w64buf[hc * Nn + n];
    float sre = 0.0f, sim = 0.0f;
    for (int j = 0; j < Jn; ++j) {
        size_t p = (size_t)j * (Bn * Hn * Nn) + idx;
        float2 in = S[p];
        S[p] = make_float2(sre, sim);
        float r = w.x * sre - w.y * sim + in.x;
        sim     = w.x * sim + w.y * sre + in.y;
        sre     = r;
    }
}

// ---------------------------------------------------------------------------
// Main fused S4 layer, one workgroup per (chunk, h), 4 waves = 64 output rows.
//   Y(64x16) = Toep_h(64x64) @ U(64x16)  +  Across_h(64x32) @ S(32x16)
// then y += D_skip*u ; h = gelu(y*gamma + beta)  (in-place on hbuf, (B,H,L))
// A fragments pre-packed in VGPR layout (one v16h load each); U tile staged
// through LDS; B fragments + skip term from LDS.
// ---------------------------------------------------------------------------
__global__ void __launch_bounds__(128)
s4_apply_kernel(float* __restrict__ hbuf,
                const v16h* __restrict__ fragC,
                const v16h* __restrict__ fragT0,
                const v16h* __restrict__ fragT1,
                const float2* __restrict__ Sb,
                const float* __restrict__ D_skip,
                int d,
                const float* __restrict__ gamma,
                const float* __restrict__ beta) {
    __shared__ float sh[Bn * LDSP];       // padded 16 x 64 tile
    int chunk = blockIdx.x;               // 0..255
    int hc    = blockIdx.y;               // 0..63
    int tid   = threadIdx.x;
    int lane  = tid & 31;
    int mi    = tid >> 5;                 // wave id = M tile (0..3)

    // ---- stage U tile: thread (b = tid>>3, q = tid&7) loads 8 floats ----
    {
        int bs = tid >> 3, q = tid & 7;
        const float4* gp = (const float4*)(hbuf + ((size_t)bs * Hn + hc) * Ln
                                           + (size_t)chunk * Tn + q * 8);
        float4 v0 = gp[0], v1 = gp[1];
        float* dst = sh + bs * LDSP + q * 8;
        dst[0] = v0.x; dst[1] = v0.y; dst[2] = v0.z; dst[3] = v0.w;
        dst[4] = v1.x; dst[5] = v1.y; dst[6] = v1.z; dst[7] = v1.w;
    }
    __syncthreads();

    // ---- A fragments: single pre-packed v16h loads ----
    int fo = (hc * 4 + mi) * 32 + lane;
    v16h a_cross = fragC[fo];
    v16h a_t0    = fragT0[fo];
    v16h a_t1    = fragT1[fo];

    // ---- B fragments from LDS / Sb ----
    int kbaseB = (lane < 16) ? 0 : 16;
    int bcol   = lane & 15;
    const float* lrow = sh + bcol * LDSP;
    v16h b_lo, b_hi, b_cross;
    #pragma unroll
    for (int j2 = 0; j2 < 16; ++j2) {
        int s = kbaseB + j2;
        b_lo[j2] = (_Float16)lrow[s];
        b_hi[j2] = (_Float16)lrow[s + 32];
    }
    const float2* sp = Sb + ((size_t)chunk * Bn + bcol) * (Hn * Nn) + (size_t)hc * Nn;
    #pragma unroll
    for (int j2 = 0; j2 < 16; ++j2) {                      // k<16: Re, k>=16: Im
        float2 v = sp[j2];
        b_cross[j2] = (_Float16)((lane < 16) ? v.x : v.y);
    }

    v8f acc = {};
    acc = __builtin_amdgcn_wmma_f32_16x16x32_f16(false, a_cross, false, b_cross,
                                                 (short)0, acc, false, false);
    acc = __builtin_amdgcn_wmma_f32_16x16x32_f16(false, a_t0, false, b_lo,
                                                 (short)0, acc, false, false);
    acc = __builtin_amdgcn_wmma_f32_16x16x32_f16(false, a_t1, false, b_hi,
                                                 (short)0, acc, false, false);

    // epilogue: skip + FiLM + gelu, vectorized in-place store (rows contiguous)
    float dsk = D_skip[d * Hn + hc];
    float g   = gamma[bcol * Hn + hc];
    float be  = beta[bcol * Hn + hc];
    int mofs = (lane < 16) ? 0 : 8;                        // C/D row layout
    int tbase = mi * 16 + mofs;
    float res[8];
    #pragma unroll
    for (int r = 0; r < 8; ++r) {
        float u = lrow[tbase + r];
        float y = acc[r] + dsk * u;
        res[r] = gelu_f(y * g + be);
    }
    float* dst = hbuf + ((size_t)bcol * Hn + hc) * Ln + (size_t)chunk * Tn + tbase;
    *(float4*)(dst)     = make_float4(res[0], res[1], res[2], res[3]);
    *(float4*)(dst + 4) = make_float4(res[4], res[5], res[6], res[7]);
}

// ---------------------------------------------------------------------------
// out[b,l] = sum_c h[b,c,l] * con_W[c] + con_b    ((B,H,L): coalesced in l)
// ---------------------------------------------------------------------------
__global__ void contract_kernel(const float* __restrict__ hbuf,
                                const float* __restrict__ con_W,
                                const float* __restrict__ con_b,
                                float* __restrict__ out) {
    size_t idx = (size_t)blockIdx.x * blockDim.x + threadIdx.x; // B*L
    int l = (int)(idx & (Ln - 1));
    int b = (int)(idx >> 14);
    const float* hp = hbuf + (size_t)b * Hn * Ln + l;
    float a = con_b[0];
    #pragma unroll 4
    for (int c = 0; c < Hn; ++c) {
        if ((c & 7) == 0 && c + 8 < Hn)
            __builtin_prefetch(hp + (size_t)(c + 8) * Ln, 0, 1);   // global_prefetch_b8
        a += hp[(size_t)c * Ln] * con_W[c];
    }
    out[idx] = a;
}

// ---------------------------------------------------------------------------
extern "C" void kernel_launch(void* const* d_in, const int* in_sizes, int n_in,
                              void* d_out, int out_size, void* d_ws, size_t ws_size,
                              hipStream_t stream) {
    const float* x      = (const float*)d_in[0];
    const float* cp     = (const float*)d_in[1];
    const float* mlp_W0 = (const float*)d_in[2];
    const float* mlp_b0 = (const float*)d_in[3];
    const float* mlp_W  = (const float*)d_in[4];
    const float* mlp_b  = (const float*)d_in[5];
    const float* exp_W  = (const float*)d_in[6];
    const float* exp_b  = (const float*)d_in[7];
    const float* con_W  = (const float*)d_in[8];
    const float* con_b  = (const float*)d_in[9];
    const float* log_dt = (const float*)d_in[10];
    const float* A_real = (const float*)d_in[11];
    const float* A_imag = (const float*)d_in[12];
    const float* C_re   = (const float*)d_in[13];
    const float* C_im   = (const float*)d_in[14];
    const float* D_skip = (const float*)d_in[15];
    const float* film_W = (const float*)d_in[16];
    const float* film_b = (const float*)d_in[17];
    float* out = (float*)d_out;

    // Workspace carve-up (all sizes multiples of 256B)
    char* ws = (char*)d_ws;
    size_t off = 0;
    float*    hbuf   = (float*)(ws + off);    off += (size_t)Bn * Ln * Hn * 4;      // 64 MiB
    float2*   Sin    = (float2*)(ws + off);   off += (size_t)Jn * Bn * Hn * Nn * 8; // 32 MiB
    float*    condb  = (float*)(ws + off);    off += Bn * MLP_HIDn * 4;
    float*    gammab = (float*)(ws + off);    off += Bn * Hn * 4;
    float*    betab  = (float*)(ws + off);    off += Bn * Hn * 4;
    float2*   wbuf   = (float2*)(ws + off);   off += Hn * Nn * 8;
    float2*   w64buf = (float2*)(ws + off);   off += Hn * Nn * 8;
    _Float16* Kf     = (_Float16*)(ws + off); off += Hn * Tn * 2;
    _Float16* Across = (_Float16*)(ws + off); off += (size_t)Hn * Tn * 32 * 2;
    v16h*     fragC  = (v16h*)(ws + off);     off += (size_t)Hn * 4 * 32 * 32;      // 256 KiB
    v16h*     fragT0 = (v16h*)(ws + off);     off += (size_t)Hn * 4 * 32 * 32;
    v16h*     fragT1 = (v16h*)(ws + off);     off += (size_t)Hn * 4 * 32 * 32;
    (void)ws_size; (void)in_sizes; (void)n_in; (void)out_size;

    // 1. conditioning MLP
    cond_mlp_kernel<<<1, Bn * MLP_HIDn, 0, stream>>>(cp, mlp_W0, mlp_b0, mlp_W, mlp_b, condb);

    // 2. expand x -> h (B,H,L)
    {
        size_t total = (size_t)Bn * Ln * Hn;
        expand_kernel<<<(unsigned)(total / 256), 256, 0, stream>>>(x, exp_W, exp_b, hbuf);
    }

    // 3. four S4D layers, chunked-scan formulation
    for (int d = 0; d < DEPTHn; ++d) {
        s4_prep_kernel<<<Hn, Tn, 0, stream>>>(log_dt, A_real, A_imag, C_re, C_im, d,
                                              Kf, Across, wbuf, w64buf);
        frag_build_kernel<<<Hn, 128, 0, stream>>>(Kf, Across, fragC, fragT0, fragT1);
        film_kernel<<<(Bn * 2 * Hn + 127) / 128, 128, 0, stream>>>(condb, film_W, film_b,
                                                                   d, gammab, betab);
        chunk_summary_kernel<<<(Jn * Bn * Hn) / 256, 256, 0, stream>>>(hbuf, wbuf, Sin);
        chunk_scan_kernel<<<(Bn * Hn * Nn) / 256, 256, 0, stream>>>(w64buf, Sin);
        s4_apply_kernel<<<dim3(Jn, Hn), 128, 0, stream>>>(hbuf, fragC, fragT0, fragT1,
                                                          Sin, D_skip, d, gammab, betab);
    }

    // 4. contract h -> out (B,L)
    contract_kernel<<<(Bn * Ln) / 256, 256, 0, stream>>>(hbuf, con_W, con_b, out);
}
